// CorrectedCorrelationLossNotEfficient_65489661329953
// MI455X (gfx1250) — compile-verified, matrix operand order checked by reading, not run on hardware
//
#include <hip/hip_runtime.h>

typedef __attribute__((ext_vector_type(16))) _Float16 v16h;
typedef __attribute__((ext_vector_type(8)))  _Float16 v8h;
typedef __attribute__((ext_vector_type(8)))  float    v8f;

#define N_ROWS 8192
#define N_COLS 1024
#define ROWCHUNKS 16
#define CHUNK (N_ROWS / ROWCHUNKS)   // 512
#define SUBROW (16 * N_ROWS)         // elements between 16-row subtiles (=262144 B)

union HV16 { v16h v; struct { v8h lo, hi; } p; };

// ---------------------------------------------------------------------------
// K0: zero the scalar output (harness poisons d_out; atomics need zero).
// ---------------------------------------------------------------------------
__global__ void zero_out(float* out) {
  if (blockIdx.x == 0 && threadIdx.x == 0) out[0] = 0.0f;
}

// ---------------------------------------------------------------------------
// K1: per-column partial sums / sums of squares over 512-row chunks.
// grid (4 colblocks, 16 rowchunks, 2 matrices), 256 threads; coalesced.
// ---------------------------------------------------------------------------
__global__ void stats_partial(const float* __restrict__ X,
                              const float* __restrict__ Y,
                              float* __restrict__ psum,
                              float* __restrict__ psq) {
  int col   = blockIdx.x * 256 + threadIdx.x;
  int chunk = blockIdx.y;
  int mat   = blockIdx.z;
  const float* A = mat ? Y : X;
  int r0 = chunk * CHUNK;
  float s = 0.0f, q = 0.0f;
  for (int r = 0; r < CHUNK; ++r) {
    float v = A[(size_t)(r0 + r) * N_COLS + col];
    s += v;
    q += v * v;
  }
  int idx = (mat * ROWCHUNKS + chunk) * N_COLS + col;
  psum[idx] = s;
  psq[idx]  = q;
}

// ---------------------------------------------------------------------------
// K2: finalize mean and 1/(std+eps) per column (ddof=1, matching reference).
// ---------------------------------------------------------------------------
__global__ void stats_final(const float* __restrict__ psum,
                            const float* __restrict__ psq,
                            float* __restrict__ stats) {
  int g   = blockIdx.x * 256 + threadIdx.x;
  int mat = g >> 10;
  int col = g & 1023;
  float s = 0.0f, q = 0.0f;
  for (int c = 0; c < ROWCHUNKS; ++c) {
    int idx = (mat * ROWCHUNKS + c) * N_COLS + col;
    s += psum[idx];
    q += psq[idx];
  }
  const float n  = (float)N_ROWS;
  float mu  = s / n;
  float var = (q - s * s / n) / (n - 1.0f);   // unbiased
  float sd  = sqrtf(fmaxf(var, 0.0f));
  stats[mat * 2048 + col]        = mu;
  stats[mat * 2048 + 1024 + col] = 1.0f / (sd + 1e-5f);
}

// ---------------------------------------------------------------------------
// K3: standardize to f16 and transpose into BhT[1024][8192] (row = column of A).
// LDS-tiled 64x64 so global reads AND writes are coalesced; LDS pitch 66
// halves keeps both phases bank-conflict-free.
// ---------------------------------------------------------------------------
__global__ void std_transpose(const float* __restrict__ X,
                              const float* __restrict__ Y,
                              const float* __restrict__ stats,
                              _Float16* __restrict__ bhtX,
                              _Float16* __restrict__ bhtY) {
  __shared__ _Float16 tile[64][66];
  int c0  = blockIdx.x * 64;
  int r0  = blockIdx.y * 64;
  int mat = blockIdx.z;
  const float* A  = mat ? Y : X;
  _Float16*   Bt  = mat ? bhtY : bhtX;
  const float* mu = stats + mat * 2048;
  const float* rs = mu + 1024;

  int tx = threadIdx.x;
  int cl = tx & 63, rg = tx >> 6;
  float m = mu[c0 + cl];
  float r = rs[c0 + cl];
#pragma unroll
  for (int i = 0; i < 16; ++i) {
    int rl = rg + i * 4;
    float v = A[(size_t)(r0 + rl) * N_COLS + c0 + cl];
    tile[cl][rl] = (_Float16)((v - m) * r);
  }
  __syncthreads();
  int rl = tx & 63, cg = tx >> 6;
#pragma unroll
  for (int i = 0; i < 16; ++i) {
    int c = cg + i * 4;
    Bt[(size_t)(c0 + c) * N_ROWS + r0 + rl] = tile[c][rl];
  }
}

// ---------------------------------------------------------------------------
// K4: fused upper-triangular SYRK-difference + masked squared-error reduce.
//
// Block tile 64(M) x 128(N); 4 waves, wave wn covers N columns
// [tj*128 + wn*32, +32). Each wave computes a 64x32 tile of BOTH C_x and
// C_y: 4 M-subtiles x 2 N-subtiles x 2 matrices = 16
// v_wmma_f32_16x16x32_f16 per 32-wide k-step from 12 subtile loads
// (43 FLOP/byte of register traffic; B operands register-reused across the
// 4 M-subtiles, A panel shared by all waves through L0).
//
// Only FOUR 64-bit base pointers are kept live (X/Y x A/B); the 16-row
// subtile displacement (s*262144 B) and the lo/hi K chunk (+32 B) are
// compile-time immediate offsets on global_load_b128, so per-k-step address
// arithmetic is just four v_add_nc_u64.
//
// Tile grid: ti in [0,16) over 64-row M-tiles, tj in [0,8) over 128-col
// N-tiles; a block exists iff some element has gm<gn, i.e. ti <= 2*tj+1
// -> 72 blocks.
// ---------------------------------------------------------------------------
__global__ void __launch_bounds__(128)
corr_gemm_loss(const _Float16* __restrict__ bhtX,
               const _Float16* __restrict__ bhtY,
               float* __restrict__ out) {
  // blockIdx.x -> (ti, tj) with ti <= 2*tj+1 (scalar loop, 8 iterations max)
  int rem = blockIdx.x;
  int tj = 0;
  while (rem >= 2 * tj + 2) { rem -= 2 * tj + 2; ++tj; }
  int ti = rem;

  int tx   = threadIdx.x;
  int lane = tx & 31;
  int wn   = tx >> 5;                 // wave id = N quarter

  int mrow0 = ti * 64;                // A operand rows (M)
  int nrow0 = tj * 128 + wn * 32;     // B operand rows (N)
  int lr    = lane & 15;
  int koff  = (lane >> 4) << 3;       // lanes 16-31 start at K=8

  const size_t yoff = (size_t)N_COLS * N_ROWS;  // BhtY = BhtX + yoff
  const _Float16* paX = bhtX + (size_t)(mrow0 + lr) * N_ROWS + koff;
  const _Float16* pbX = bhtX + (size_t)(nrow0 + lr) * N_ROWS + koff;
  const _Float16* paY = paX + yoff;
  const _Float16* pbY = pbX + yoff;

  v8f accX[4][2] = {};   // [m-sub][n-sub]
  v8f accY[4][2] = {};

  for (int k0 = 0; k0 < N_ROWS; k0 += 32) {
    HV16 aX[4], aY[4], bX[2], bY[2];
#pragma unroll
    for (int s = 0; s < 4; ++s) {
      aX[s].p.lo = *reinterpret_cast<const v8h*>(paX + k0 + s * SUBROW);
      aX[s].p.hi = *reinterpret_cast<const v8h*>(paX + k0 + s * SUBROW + 16);
      aY[s].p.lo = *reinterpret_cast<const v8h*>(paY + k0 + s * SUBROW);
      aY[s].p.hi = *reinterpret_cast<const v8h*>(paY + k0 + s * SUBROW + 16);
    }
#pragma unroll
    for (int t = 0; t < 2; ++t) {
      bX[t].p.lo = *reinterpret_cast<const v8h*>(pbX + k0 + t * SUBROW);
      bX[t].p.hi = *reinterpret_cast<const v8h*>(pbX + k0 + t * SUBROW + 16);
      bY[t].p.lo = *reinterpret_cast<const v8h*>(pbY + k0 + t * SUBROW);
      bY[t].p.hi = *reinterpret_cast<const v8h*>(pbY + k0 + t * SUBROW + 16);
    }
#pragma unroll
    for (int s = 0; s < 4; ++s)
#pragma unroll
      for (int t = 0; t < 2; ++t) {
        accX[s][t] = __builtin_amdgcn_wmma_f32_16x16x32_f16(
            false, aX[s].v, false, bX[t].v, (short)0, accX[s][t], false, false);
        accY[s][t] = __builtin_amdgcn_wmma_f32_16x16x32_f16(
            false, aY[s].v, false, bY[t].v, (short)0, accY[s][t], false, false);
      }
  }

  // masked squared difference on the accumulator layout:
  // element e, lane l -> M = e + 8*(l>=16), N = l&15
  float local = 0.0f;
  int baseM = mrow0 + ((lane >> 4) << 3);
  int baseN = nrow0 + (lane & 15);
#pragma unroll
  for (int s = 0; s < 4; ++s)
#pragma unroll
    for (int t = 0; t < 2; ++t)
#pragma unroll
      for (int e = 0; e < 8; ++e) {
        int gm = baseM + s * 16 + e;
        int gn = baseN + t * 16;
        if (gm < gn) {
          float d = accX[s][t][e] - accY[s][t][e];
          local += d * d;
        }
      }

  __shared__ float red[128];
  red[tx] = local;
  __syncthreads();
  if (tx < 64) red[tx] += red[tx + 64];
  __syncthreads();
  if (tx < 32) red[tx] += red[tx + 32];
  __syncthreads();
  if (tx == 0) {
    float s = 0.0f;
#pragma unroll
    for (int i = 0; i < 32; ++i) s += red[i];
    const float n     = (float)N_ROWS;
    const float count = (float)((N_COLS * (N_COLS - 1)) / 2);
    // accumulators hold n * C entries; fold 1/n^2 and 1/count here
    atomicAdd(out, s / (n * n) / count);
  }
}

// ---------------------------------------------------------------------------
extern "C" void kernel_launch(void* const* d_in, const int* in_sizes, int n_in,
                              void* d_out, int out_size, void* d_ws, size_t ws_size,
                              hipStream_t stream) {
  const float* X = (const float*)d_in[0];
  const float* Y = (const float*)d_in[1];
  float* out = (float*)d_out;

  // Workspace layout (floats): psum[32768] | psq[32768] | stats[4096] | BhT f16
  float* ws    = (float*)d_ws;
  float* psum  = ws;
  float* psq   = ws + 32768;
  float* stats = ws + 65536;
  _Float16* bht  = (_Float16*)(ws + 69632);   // 16B-aligned offset
  _Float16* bhtX = bht;
  _Float16* bhtY = bht + (size_t)N_COLS * N_ROWS;

  zero_out<<<1, 64, 0, stream>>>(out);
  stats_partial<<<dim3(4, ROWCHUNKS, 2), 256, 0, stream>>>(X, Y, psum, psq);
  stats_final<<<8, 256, 0, stream>>>(psum, psq, stats);
  std_transpose<<<dim3(16, 128, 2), 256, 0, stream>>>(X, Y, stats, bhtX, bhtY);
  corr_gemm_loss<<<72, 128, 0, stream>>>(bhtX, bhtY, out);
}
